// BertSelfAttention_214748365264
// MI455X (gfx1250) — matrix operand order repaired and checked by reading, not compiled, not run on hardware
//
#include <hip/hip_runtime.h>

// ---------------------------------------------------------------------------
// BERT self-attention forward for gfx1250 (MI455X), bf16 WMMA path.
// B=8, S=1024, HID=1024, NH=16, HD=64.
// v2: 32-row wave tiles in both GEMM and attention to double WMMA work per
//     loaded B-fragment (compute-bound; data lives in the 192MB L2).
// ---------------------------------------------------------------------------

typedef __attribute__((ext_vector_type(16))) __bf16 v16bf;
typedef __attribute__((ext_vector_type(8)))  float  v8f;

#define BATCH 8
#define SEQ   1024
#define HID   1024
#define NH    16
#define HD    64
#define ROWS  (BATCH * SEQ)   // 8192

__device__ __forceinline__ unsigned short f2bf(float f) {
    unsigned int u = __float_as_uint(f);
    u += 0x7FFFu + ((u >> 16) & 1u);   // round-to-nearest-even
    return (unsigned short)(u >> 16);
}

union FragU { uint4 q[2]; v16bf v; };

// A-fragment: 16x32 bf16 tile, source row-major with leading dim ld.
// Lane L -> M = L%16; per ISA 16-bit A layout: VGPR0-3 = K(k0 + half*8 + 0..7),
// VGPR4-7 = K(k0 + 16 + half*8 + 0..7). Two aligned 128-bit loads.
__device__ __forceinline__ v16bf load_a_frag(const unsigned short* p, int ld,
                                             int row, int k0, int lane) {
    const int half = (lane >> 4) & 1;
    const unsigned short* base = p + (size_t)row * ld + (k0 + half * 8);
    FragU f;
    f.q[0] = *reinterpret_cast<const uint4*>(base);
    f.q[1] = *reinterpret_cast<const uint4*>(base + 16);
    return f.v;
}

// B-fragment: 32x16 bf16 tile. Source stores each B-column (N index) as a
// contiguous K-run ("N-major", leading dim ld). Lane L -> N = L%16;
// lane half h holds K=k0+h*8+0..7 then K=k0+16+h*8+0..7 (matches A layout).
__device__ __forceinline__ v16bf load_b_frag(const unsigned short* p, int ld,
                                             int nrow, int k0, int lane) {
    const int half = (lane >> 4) & 1;
    const unsigned short* base = p + (size_t)nrow * ld + (k0 + half * 8);
    FragU f;
    f.q[0] = *reinterpret_cast<const uint4*>(base);
    f.q[1] = *reinterpret_cast<const uint4*>(base + 16);
    return f.v;
}

__device__ __forceinline__ v8f wmma_bf16(v16bf a, v16bf b, v8f c) {
    return __builtin_amdgcn_wmma_f32_16x16x32_bf16(
        /*neg_a=*/false, a, /*neg_b=*/false, b,
        /*c_mod=*/(short)0, c, /*reuse_a=*/false, /*reuse_b=*/false);
}

// --------------------------- conversion kernels ----------------------------

__global__ void k_f32_to_bf16(const float* __restrict__ in,
                              unsigned short* __restrict__ out, int n) {
    int i = blockIdx.x * blockDim.x + threadIdx.x;
    if (i < n) out[i] = f2bf(in[i]);
}

// W [K=1024][N=1024] fp32 row-major -> Wt [N][K] bf16 (so B-frags are row reads)
__global__ void k_w_transpose_bf16(const float* __restrict__ W,
                                   unsigned short* __restrict__ Wt) {
    int i = blockIdx.x * blockDim.x + threadIdx.x;   // 0 .. 1M-1
    int k = i >> 10;
    int n = i & 1023;
    Wt[(size_t)n * HID + k] = f2bf(W[i]);
}

// ------------------------------- QKV GEMM ----------------------------------
// One wave computes a 32(M) x 64(N) output tile of X[8192,1024] @ W[1024,1024]
// + bias: per k-step 2 A-frags + 4 B-frags -> 8 WMMAs (B-frags reused 2x).
__global__ void k_qkv_gemm(const unsigned short* __restrict__ Xb,
                           const unsigned short* __restrict__ Wt,
                           const float* __restrict__ bias,
                           unsigned short* __restrict__ out,
                           int v_transposed) {
    const int lane = threadIdx.x & 31;
    const int gw   = blockIdx.x * (blockDim.x >> 5) + (threadIdx.x >> 5);
    const int nt   = gw & 15;         // 16 N-tiles of 64
    const int mt   = gw >> 4;         // 256 M-tiles of 32
    const int m0   = mt << 5;
    const int n0   = nt << 6;
    const int row  = lane & 15;
    const int half = lane >> 4;

    v8f acc[2][4];
    #pragma unroll
    for (int mi = 0; mi < 2; ++mi)
        #pragma unroll
        for (int t = 0; t < 4; ++t)
            #pragma unroll
            for (int r = 0; r < 8; ++r) acc[mi][t][r] = 0.0f;

    for (int k = 0; k < HID; k += 32) {
        v16bf a0 = load_a_frag(Xb, HID, m0 + row,      k, lane);
        v16bf a1 = load_a_frag(Xb, HID, m0 + 16 + row, k, lane);
        #pragma unroll
        for (int t = 0; t < 4; ++t) {
            v16bf bm = load_b_frag(Wt, HID, n0 + t * 16 + row, k, lane);
            acc[0][t] = wmma_bf16(a0, bm, acc[0][t]);
            acc[1][t] = wmma_bf16(a1, bm, acc[1][t]);
        }
    }

    const int h = n0 >> 6;            // whole wave maps to one head
    #pragma unroll
    for (int t = 0; t < 4; ++t) {
        const int   n   = n0 + t * 16 + row;
        const float bia = bias[n];
        const int   hd  = n - (h << 6);
        #pragma unroll
        for (int mi = 0; mi < 2; ++mi)
            #pragma unroll
            for (int r = 0; r < 8; ++r) {
                const int m  = m0 + mi * 16 + r + half * 8;
                const int bb = m >> 10;
                const int s  = m & 1023;
                const float val = acc[mi][t][r] + bia;
                size_t off;
                if (v_transposed)
                    off = (((size_t)(bb * NH + h) * HD + hd) << 10) + s;   // [B,NH,HD,S]
                else
                    off = ((((size_t)(bb * NH + h)) << 10) + s) * HD + hd; // [B,NH,S,HD]
                out[off] = f2bf(val);
            }
    }
}

// ------------------------------- attention ---------------------------------
// One wave handles 32 query rows of one (b,h): flash-style online softmax.
// V-fragments are shared across the two query row-blocks.
__global__ void k_attn(const unsigned short* __restrict__ Qb,
                       const unsigned short* __restrict__ Kb,
                       const unsigned short* __restrict__ Vtb,
                       const float* __restrict__ mask,
                       float* __restrict__ out) {
    __shared__ __align__(16) unsigned short pstage[8][32 * 32]; // 2KB per wave

    const int lane = threadIdx.x & 31;
    const int wib  = threadIdx.x >> 5;
    const int gw   = blockIdx.x * 8 + wib;
    const int qt   = gw & 31;          // 32 query tiles of 32
    const int bh   = gw >> 5;          // 128 (b,h) pairs
    const int b    = bh >> 4;
    const int h    = bh & 15;
    const int m0   = qt << 5;
    const int row  = lane & 15;
    const int half = lane >> 4;

    const unsigned short* Qp = Qb  + (size_t)bh * (SEQ * HD);
    const unsigned short* Kp = Kb  + (size_t)bh * (SEQ * HD);
    const unsigned short* Vp = Vtb + (size_t)bh * (HD * SEQ);
    const float*          mp = mask + (size_t)b * SEQ;

    // Q tile, 32x64 -> four A-frags [row-block][k-half], pinned in VGPRs
    v16bf qa[2][2];
    #pragma unroll
    for (int mi = 0; mi < 2; ++mi) {
        qa[mi][0] = load_a_frag(Qp, HD, m0 + mi * 16 + row, 0, lane);
        qa[mi][1] = load_a_frag(Qp, HD, m0 + mi * 16 + row, 32, lane);
    }

    v8f o[2][4], mv[2], lv[2];
    #pragma unroll
    for (int mi = 0; mi < 2; ++mi) {
        #pragma unroll
        for (int t = 0; t < 4; ++t)
            #pragma unroll
            for (int r = 0; r < 8; ++r) o[mi][t][r] = 0.0f;
        #pragma unroll
        for (int r = 0; r < 8; ++r) { mv[mi][r] = -1e30f; lv[mi][r] = 0.0f; }
    }

    for (int kc = 0; kc < SEQ; kc += 32) {
        float mval[2];
        #pragma unroll
        for (int t = 0; t < 2; ++t) mval[t] = mp[kc + t * 16 + row];

        v16bf pa[2];
        #pragma unroll
        for (int mi = 0; mi < 2; ++mi) {
            // scores: 16 queries x 32 keys, fp32, as two 16x16 frags
            v8f s[2];
            #pragma unroll
            for (int t = 0; t < 2; ++t) {
                v8f z;
                #pragma unroll
                for (int r = 0; r < 8; ++r) z[r] = 0.0f;
                v16bf kb0 = load_b_frag(Kp, HD, kc + t * 16 + row, 0, lane);
                v16bf kb1 = load_b_frag(Kp, HD, kc + t * 16 + row, 32, lane);
                s[t] = wmma_bf16(qa[mi][0], kb0, z);
                s[t] = wmma_bf16(qa[mi][1], kb1, s[t]);
                #pragma unroll
                for (int r = 0; r < 8; ++r) s[t][r] = s[t][r] * 0.125f + mval[t];
            }

            // row max (reduce across the 16-lane half holding this row)
            v8f rm;
            #pragma unroll
            for (int r = 0; r < 8; ++r) rm[r] = fmaxf(s[0][r], s[1][r]);
            for (int xm = 1; xm <= 8; xm <<= 1)
                #pragma unroll
                for (int r = 0; r < 8; ++r)
                    rm[r] = fmaxf(rm[r], __shfl_xor(rm[r], xm, 32));

            v8f mnew, sc;
            #pragma unroll
            for (int r = 0; r < 8; ++r) {
                mnew[r]    = fmaxf(mv[mi][r], rm[r]);
                sc[r]      = __expf(mv[mi][r] - mnew[r]);
                mv[mi][r]  = mnew[r];
                lv[mi][r] *= sc[r];
            }
            #pragma unroll
            for (int t = 0; t < 4; ++t)
                #pragma unroll
                for (int r = 0; r < 8; ++r) o[mi][t][r] *= sc[r];

            // P = exp(S - m), row sums
            #pragma unroll
            for (int t = 0; t < 2; ++t)
                #pragma unroll
                for (int r = 0; r < 8; ++r) s[t][r] = __expf(s[t][r] - mnew[r]);
            v8f rs;
            #pragma unroll
            for (int r = 0; r < 8; ++r) rs[r] = s[0][r] + s[1][r];
            for (int xm = 1; xm <= 8; xm <<= 1)
                #pragma unroll
                for (int r = 0; r < 8; ++r) rs[r] += __shfl_xor(rs[r], xm, 32);
            #pragma unroll
            for (int r = 0; r < 8; ++r) lv[mi][r] += rs[r];

            // C-layout -> A-layout for P via per-wave LDS tile (32x32 bf16)
            unsigned short* pl = &pstage[wib][0];
            #pragma unroll
            for (int t = 0; t < 2; ++t)
                #pragma unroll
                for (int r = 0; r < 8; ++r)
                    pl[(mi * 16 + r + half * 8) * 32 + t * 16 + row] = f2bf(s[t][r]);

            FragU pf;
            const unsigned short* pb = pl + (mi * 16 + row) * 32 + half * 8;
            pf.q[0] = *reinterpret_cast<const uint4*>(pb);
            pf.q[1] = *reinterpret_cast<const uint4*>(pb + 16);
            pa[mi] = pf.v;
        }

        // O += P @ V : each V-fragment feeds both query row-blocks
        #pragma unroll
        for (int t2 = 0; t2 < 4; ++t2) {
            v16bf vb = load_b_frag(Vp, SEQ, t2 * 16 + row, kc, lane);
            o[0][t2] = wmma_bf16(pa[0], vb, o[0][t2]);
            o[1][t2] = wmma_bf16(pa[1], vb, o[1][t2]);
        }
    }

    // normalize & store to [B,S,HID] fp32
    #pragma unroll
    for (int mi = 0; mi < 2; ++mi) {
        v8f rl;
        #pragma unroll
        for (int r = 0; r < 8; ++r) rl[r] = 1.0f / lv[mi][r];
        #pragma unroll
        for (int t2 = 0; t2 < 4; ++t2)
            #pragma unroll
            for (int r = 0; r < 8; ++r) {
                const int s_idx = m0 + mi * 16 + r + half * 8;
                const int col   = h * HD + t2 * 16 + row;
                out[(((size_t)b << 10) + s_idx) * HID + col] = o[mi][t2][r] * rl[r];
            }
    }
}

// ------------------------------ host launch --------------------------------

extern "C" void kernel_launch(void* const* d_in, const int* in_sizes, int n_in,
                              void* d_out, int out_size, void* d_ws, size_t ws_size,
                              hipStream_t stream) {
    const float* hs   = (const float*)d_in[0];
    const float* mask = (const float*)d_in[1];
    const float* Wq   = (const float*)d_in[2];
    const float* bq   = (const float*)d_in[3];
    const float* Wk   = (const float*)d_in[4];
    const float* bk   = (const float*)d_in[5];
    const float* Wv   = (const float*)d_in[6];
    const float* bv   = (const float*)d_in[7];
    float*       out  = (float*)d_out;

    char* ws = (char*)d_ws;
    const size_t SZ_X  = (size_t)ROWS * HID * 2;            // 16 MB
    const size_t SZ_W  = (size_t)HID * HID * 2;             //  2 MB each
    const size_t SZ_T  = (size_t)BATCH * NH * SEQ * HD * 2; // 16 MB each
    unsigned short* Xb  = (unsigned short*)(ws);
    unsigned short* Wqt = (unsigned short*)(ws + SZ_X);
    unsigned short* Wkt = (unsigned short*)(ws + SZ_X + SZ_W);
    unsigned short* Wvt = (unsigned short*)(ws + SZ_X + 2 * SZ_W);
    unsigned short* Qb  = (unsigned short*)(ws + SZ_X + 3 * SZ_W);
    unsigned short* Kb  = (unsigned short*)(ws + SZ_X + 3 * SZ_W + SZ_T);
    unsigned short* Vtb = (unsigned short*)(ws + SZ_X + 3 * SZ_W + 2 * SZ_T);

    // 1) fp32 -> bf16 conversions
    k_f32_to_bf16<<<(ROWS * HID) / 256, 256, 0, stream>>>(hs, Xb, ROWS * HID);
    k_w_transpose_bf16<<<(HID * HID) / 256, 256, 0, stream>>>(Wq, Wqt);
    k_w_transpose_bf16<<<(HID * HID) / 256, 256, 0, stream>>>(Wk, Wkt);
    k_w_transpose_bf16<<<(HID * HID) / 256, 256, 0, stream>>>(Wv, Wvt);

    // 2) QKV projections (4096 waves each; 8 waves / 256-thread block)
    const int gemm_blocks = (ROWS / 32) * (HID / 64) / 8;   // 512
    k_qkv_gemm<<<gemm_blocks, 256, 0, stream>>>(Xb, Wqt, bq, Qb, 0);
    k_qkv_gemm<<<gemm_blocks, 256, 0, stream>>>(Xb, Wkt, bk, Kb, 0);
    k_qkv_gemm<<<gemm_blocks, 256, 0, stream>>>(Xb, Wvt, bv, Vtb, 1);

    // 3) fused softmax-attention (4096 waves: B*NH*S/32)
    const int attn_blocks = (BATCH * NH * (SEQ / 32)) / 8;  // 512
    k_attn<<<attn_blocks, 256, 0, stream>>>(Qb, Kb, Vtb, mask, out);
}
